// DGIEncoderNet_49289044689577
// MI455X (gfx1250) — compile-verified, matrix operand order checked by reading, not run on hardware
//
#include <hip/hip_runtime.h>

#define NN 100000
#define NE 3200000

typedef __attribute__((ext_vector_type(16))) _Float16 v16h;
typedef __attribute__((ext_vector_type(8)))  _Float16 v8h;
typedef __attribute__((ext_vector_type(8)))  float    v8f;

// ---------------- norm pipeline ----------------

__global__ void k_fill(float* __restrict__ p, float v, int n) {
  int i = blockIdx.x * blockDim.x + threadIdx.x;
  if (i < n) p[i] = v;
}

__global__ void k_deg(const long long* __restrict__ col, const float* __restrict__ ea,
                      float* __restrict__ deg, int e) {
  int i = blockIdx.x * blockDim.x + threadIdx.x;
  if (i < e) unsafeAtomicAdd(&deg[(int)col[i]], ea[i]);
}

__global__ void k_dinv(const float* __restrict__ deg, float* __restrict__ dinv, int n) {
  int i = blockIdx.x * blockDim.x + threadIdx.x;
  if (i < n) {
    float d = deg[i];
    dinv[i] = d > 0.f ? rsqrtf(fmaxf(d, 1e-30f)) : 0.f;
  }
}

__global__ void k_edgenorm(const long long* __restrict__ row, const long long* __restrict__ col,
                           const float* __restrict__ ea, const float* __restrict__ dinv,
                           float* __restrict__ nrm, int e) {
  int i = blockIdx.x * blockDim.x + threadIdx.x;
  if (i < e) nrm[i] = dinv[(int)row[i]] * ea[i] * dinv[(int)col[i]];
}

// ---------------- weight repack into WMMA B-fragment order ----------------
// B frag (16-bit, 32xK x 16): lane L holds column n = ct*16 + (L&15);
// K values: base = (L<16?0:8); elem e<8 -> K = kc*32+base+e ; e>=8 -> K = kc*32+base+16+(e-8)
__global__ void k_pack(const float* __restrict__ W, _Float16* __restrict__ Bp, int FIN, int FOUT) {
  int idx = blockIdx.x * blockDim.x + threadIdx.x;
  if (idx >= FIN * FOUT) return;
  int e    = idx & 15;
  int lane = (idx >> 4) & 31;
  int frag = idx >> 9;
  int CT = FOUT / 16;
  int kc = frag / CT, ct = frag - kc * CT;
  int base = (lane >> 4) * 8;
  int k = kc * 32 + base + (e < 8 ? e : 16 + (e - 8));
  int n = ct * 16 + (lane & 15);
  Bp[idx] = (_Float16)W[(size_t)k * FOUT + n];
}

// ---------------- f32 -> f16 (optionally ELU first) ----------------

__global__ void k_cvt(const float* __restrict__ in, _Float16* __restrict__ out, long long n) {
  long long i = (long long)blockIdx.x * blockDim.x + threadIdx.x;
  if (i < n) out[i] = (_Float16)in[i];
}

__global__ void k_elu_cvt(const float* __restrict__ in, _Float16* __restrict__ out, long long n) {
  long long i = (long long)blockIdx.x * blockDim.x + threadIdx.x;
  if (i < n) {
    float v = in[i];
    v = v > 0.f ? v : (__expf(v) - 1.f);
    out[i] = (_Float16)v;
  }
}

// ---------------- WMMA GEMM: H[N,FOUT] = Xh[N,FIN] @ W  ----------------
// One wave computes a 16-row tile. N_NODES = 6250 * 16 exactly, grid is exact,
// so EXEC is all-ones throughout (WMMA requirement).
template <int FIN, int FOUT>
__global__ void k_gemm(const _Float16* __restrict__ X, const _Float16* __restrict__ Bp,
                       float* __restrict__ H) {
  constexpr int KCH = FIN / 32;
  constexpr int CT  = FOUT / 16;
  int wave = blockIdx.x * 2 + (threadIdx.x >> 5);
  int lane = threadIdx.x & 31;
  int half = lane >> 4;
  int l16  = lane & 15;
  int base = half * 8;
  int row  = wave * 16 + l16;

  // A fragments: lane holds row M=l16, K = base+{0..7} and base+16+{0..7} per chunk
  v16h afr[KCH];
  const _Float16* xr = X + (size_t)row * FIN;
#pragma unroll
  for (int kc = 0; kc < KCH; ++kc) {
    v8h lo = *(const v8h*)(xr + kc * 32 + base);
    v8h hi = *(const v8h*)(xr + kc * 32 + base + 16);
#pragma unroll
    for (int i = 0; i < 8; ++i) { afr[kc][i] = lo[i]; afr[kc][8 + i] = hi[i]; }
  }

#pragma unroll
  for (int ct = 0; ct < CT; ++ct) {
    v8f c = {};
#pragma unroll
    for (int kc = 0; kc < KCH; ++kc) {
      v16h b = *(const v16h*)(Bp + ((size_t)(kc * CT + ct) * 32 + lane) * 16);
      c = __builtin_amdgcn_wmma_f32_16x16x32_f16(false, afr[kc], false, b,
                                                 (short)0, c, false, false);
    }
    // D layout: VGPR i -> row = tile*16 + i + half*8, col = ct*16 + l16
    float* ho = H + (size_t)(wave * 16 + half * 8) * FOUT + ct * 16 + l16;
#pragma unroll
    for (int i = 0; i < 8; ++i) ho[(size_t)i * FOUT] = c[i];
  }
}

// ---------------- aggregation: out = dinv^2 * h + b (self-loop term + bias) -------------

__global__ void k_aggr_init(const float* __restrict__ H, const float* __restrict__ dinv,
                            const float* __restrict__ b, float* __restrict__ Out,
                            int lg, int fmask, long long total) {
  long long i = (long long)blockIdx.x * blockDim.x + threadIdx.x;
  if (i < total) {
    int n = (int)(i >> lg);
    int f = (int)i & fmask;
    float s = dinv[n];
    Out[i] = s * s * H[i] + b[f];
  }
}

// ---------------- edge scatter: Out[col] += norm * H[row]  ----------------
// FOUT/4 threads per edge, float4 gathers (coalesced 128B+ per edge), hw f32 atomics.
template <int FOUT>
__global__ void k_scatter(const long long* __restrict__ row, const long long* __restrict__ col,
                          const float* __restrict__ nrm, const float* __restrict__ H,
                          float* __restrict__ Out) {
  constexpr int TPE = FOUT / 4;
  long long gid = (long long)blockIdx.x * blockDim.x + threadIdx.x;
  long long e = gid / TPE;
  if (e >= NE) return;
  int f = (int)(gid - e * TPE) * 4;
  int r = (int)row[e];
  int c = (int)col[e];
  float w = nrm[e];
  float4 v = *(const float4*)(H + (size_t)r * FOUT + f);
  float* o = Out + (size_t)c * FOUT + f;
  unsafeAtomicAdd(o + 0, w * v.x);
  unsafeAtomicAdd(o + 1, w * v.y);
  unsafeAtomicAdd(o + 2, w * v.z);
  unsafeAtomicAdd(o + 3, w * v.w);
}

// ---------------- host orchestration ----------------

extern "C" void kernel_launch(void* const* d_in, const int* in_sizes, int n_in,
                              void* d_out, int out_size, void* d_ws, size_t ws_size,
                              hipStream_t stream) {
  const float*     x   = (const float*)d_in[0];
  const long long* ei  = (const long long*)d_in[1];
  const long long* rowv = ei;            // edge_index[0] = source
  const long long* colv = ei + NE;       // edge_index[1] = target
  const float*     ea  = (const float*)d_in[2];
  const float*     W1  = (const float*)d_in[3];
  const float*     b1  = (const float*)d_in[4];
  const float*     W2  = (const float*)d_in[5];
  const float*     b2  = (const float*)d_in[6];
  const float*     W3  = (const float*)d_in[7];
  const float*     b3  = (const float*)d_in[8];
  float*           out = (float*)d_out;

  // workspace carve (256B aligned)
  char* base = (char*)d_ws;
  size_t off = 0;
  auto alloc = [&](size_t bytes) -> void* {
    void* p = base + off;
    off += (bytes + 255) & ~(size_t)255;
    return p;
  };
  float*     deg  = (float*)alloc((size_t)NN * 4);
  float*     dinv = (float*)alloc((size_t)NN * 4);
  float*     nrm  = (float*)alloc((size_t)NE * 4);
  _Float16*  Bp1  = (_Float16*)alloc((size_t)128 * 32 * 2);
  _Float16*  Bp2  = (_Float16*)alloc((size_t)32 * 64 * 2);
  _Float16*  Bp3  = (_Float16*)alloc((size_t)64 * 128 * 2);
  _Float16*  xh   = (_Float16*)alloc((size_t)NN * 128 * 2);  // f16 activations (reused)
  float*     h    = (float*)alloc((size_t)NN * 128 * 4);     // GEMM output (reused)
  float*     agg  = (float*)alloc((size_t)NN * 64 * 4);      // layer-1/2 aggregation

  const int T = 256;
  auto nb = [&](long long n) { return (unsigned)((n + T - 1) / T); };

  // --- gcn_norm ---
  k_fill<<<nb(NN), T, 0, stream>>>(deg, 1.0f, NN);                 // self-loop weight
  k_deg<<<nb(NE), T, 0, stream>>>(colv, ea, deg, NE);
  k_dinv<<<nb(NN), T, 0, stream>>>(deg, dinv, NN);
  k_edgenorm<<<nb(NE), T, 0, stream>>>(rowv, colv, ea, dinv, nrm, NE);

  // --- pack weights into WMMA B-fragment order ---
  k_pack<<<nb(128 * 32), T, 0, stream>>>(W1, Bp1, 128, 32);
  k_pack<<<nb(32 * 64), T, 0, stream>>>(W2, Bp2, 32, 64);
  k_pack<<<nb(64 * 128), T, 0, stream>>>(W3, Bp3, 64, 128);

  const unsigned GEMM_BLOCKS = NN / 16 / 2;  // 6250 tiles, 2 waves/block (exact)

  // --- layer 1: 128 -> 32, ELU ---
  long long nx = (long long)NN * 128;
  k_cvt<<<nb(nx), T, 0, stream>>>(x, xh, nx);
  k_gemm<128, 32><<<GEMM_BLOCKS, 64, 0, stream>>>(xh, Bp1, h);
  long long t1 = (long long)NN * 32;
  k_aggr_init<<<nb(t1), T, 0, stream>>>(h, dinv, b1, agg, 5, 31, t1);
  k_scatter<32><<<nb((long long)NE * 8), T, 0, stream>>>(rowv, colv, nrm, h, agg);
  k_elu_cvt<<<nb(t1), T, 0, stream>>>(agg, xh, t1);

  // --- layer 2: 32 -> 64, ELU ---
  k_gemm<32, 64><<<GEMM_BLOCKS, 64, 0, stream>>>(xh, Bp2, h);
  long long t2 = (long long)NN * 64;
  k_aggr_init<<<nb(t2), T, 0, stream>>>(h, dinv, b2, agg, 6, 63, t2);
  k_scatter<64><<<nb((long long)NE * 16), T, 0, stream>>>(rowv, colv, nrm, h, agg);
  k_elu_cvt<<<nb(t2), T, 0, stream>>>(agg, xh, t2);

  // --- layer 3: 64 -> 128, no ELU, straight into d_out ---
  k_gemm<64, 128><<<GEMM_BLOCKS, 64, 0, stream>>>(xh, Bp3, h);
  long long t3 = (long long)NN * 128;
  k_aggr_init<<<nb(t3), T, 0, stream>>>(h, dinv, b3, out, 7, 127, t3);
  k_scatter<128><<<nb((long long)NE * 32), T, 0, stream>>>(rowv, colv, nrm, h, out);
}
